// GatingAttentionLayerWsa_5016521802186
// MI455X (gfx1250) — matrix operand order repaired and checked by reading, not compiled, not run on hardware
//
#include <hip/hip_runtime.h>
#include <hip/hip_bf16.h>

// ---------------------------------------------------------------------------
// GatingAttentionLayerWsa for MI455X (gfx1250, wave32, WMMA bf16 16x16x32)
//   B=4, S=2048, E=512, H=8, D=64
// Pipeline: 3x WMMA proj GEMMs (fp32->bf16) -> fused 2-pass gated attention
//           (TDM tensor_load_to_lds tile staging, row mean/var norm + softmax
//            + P.V in LDS) -> WMMA out GEMM.
// Workspace: qp/kp/vp/ao bf16, 4 * 8192*512*2 B = 33.5 MB.
// ---------------------------------------------------------------------------

typedef __attribute__((ext_vector_type(16))) __bf16 v16bf;
typedef __attribute__((ext_vector_type(4)))  __bf16 v4bf;
typedef __attribute__((ext_vector_type(8)))  float  v8f;
typedef __attribute__((ext_vector_type(4)))  unsigned int v4u;
typedef __attribute__((ext_vector_type(8)))  int    v8i;
typedef __attribute__((ext_vector_type(4)))  int    v4i;

#define WMMA_BF16(a, b, c) \
  __builtin_amdgcn_wmma_f32_16x16x32_bf16(false, (a), false, (b), (short)0, (c), false, false)

#if __has_builtin(__builtin_amdgcn_tensor_load_to_lds)
#define USE_TDM 1
#else
#define USE_TDM 0
#endif

#if __has_builtin(__builtin_amdgcn_s_wait_tensorcnt)
#define TDM_WAIT() do { __builtin_amdgcn_s_wait_tensorcnt((short)0); \
                        asm volatile("" ::: "memory"); } while (0)
#else
#define TDM_WAIT() asm volatile("s_wait_tensorcnt 0x0" ::: "memory")
#endif

static constexpr int Bb = 4, Ss = 2048, Ee = 512, Hh = 8, Dd = 64;
static constexpr int Mrows = Bb * Ss;      // 8192
static constexpr int Ncols = Hh * Dd;      // 512

// ---- fragment loaders (contiguous k = 16h + s, same for A and B) ------------------
// A: 16x32 bf16. lanes 0-15: K 0..15 ; lanes 16-31: K 16..31.
__device__ __forceinline__ v16bf ldsA_frag(const __bf16* A, int ld) {
  const int lane = threadIdx.x & 31;
  const int m  = lane & 15;
  const int kb = (lane < 16) ? 0 : 16;
  v16bf a;
#pragma unroll
  for (int i = 0; i < 4; ++i) {
    const int k = kb + 2 * i;
    a[2 * i]     = A[m * ld + k];
    a[2 * i + 1] = A[m * ld + k + 1];
    a[2 * i + 8] = A[m * ld + k + 8];
    a[2 * i + 9] = A[m * ld + k + 9];
  }
  return a;
}

// B: 32x16 bf16 stored row-major [k][n], row stride ld.
__device__ __forceinline__ v16bf ldsB_frag(const __bf16* Bm, int ld) {
  const int lane = threadIdx.x & 31;
  const int n  = lane & 15;
  const int kb = (lane < 16) ? 0 : 16;
  v16bf b;
#pragma unroll
  for (int i = 0; i < 8; ++i) {
    b[2 * i]     = Bm[(kb + 2 * i) * ld + n];
    b[2 * i + 1] = Bm[(kb + 2 * i + 1) * ld + n];
  }
  return b;
}

// B from transposed storage: logical B[k][n] = T[n*ld + k]
__device__ __forceinline__ v16bf ldsBT_frag(const __bf16* T, int ld) {
  const int lane = threadIdx.x & 31;
  const int n  = lane & 15;
  const int kb = (lane < 16) ? 0 : 16;
  v16bf b;
#pragma unroll
  for (int i = 0; i < 8; ++i) {
    b[2 * i]     = T[n * ld + kb + 2 * i];
    b[2 * i + 1] = T[n * ld + kb + 2 * i + 1];
  }
  return b;
}

__device__ __forceinline__ float half_reduce(float v) {
  v += __shfl_xor(v, 1, 32);
  v += __shfl_xor(v, 2, 32);
  v += __shfl_xor(v, 4, 32);
  v += __shfl_xor(v, 8, 32);
  return v;
}

#if USE_TDM
// LDS byte offset of a __shared__ object (generic ptr low 32 bits = LDS addr)
__device__ __forceinline__ unsigned lds_off(const void* p) {
  return (unsigned)(size_t)p;
}

// TDM D# for a 2-D bf16 tile: `rows` x 64 elements, global row stride 512
// elems, LDS destination padded 32 dwords -> +4 dwords (row pitch 72 bf16).
// ISA 08_async_tensor §8: group0 = {flags,count | lds_addr | gaddr57 | type=2},
// group1 = {mask/dsize/pad | dims | tile dims | strides}. Groups 2/3 zero (2D).
// This toolchain's builtin takes 6 args (g0, g1, g2, g3, g4, cpol).
__device__ __forceinline__ void tdm_load_tile(const __bf16* gsrc,
                                              unsigned lds_addr, int rows) {
  const unsigned long long ga = (unsigned long long)(size_t)gsrc;
  v4u g0;
  g0[0] = 1u;                                           // count=1, user mode
  g0[1] = lds_addr;                                     // lds_addr
  g0[2] = (unsigned)(ga & 0xffffffffu);                 // global_addr[31:0]
  g0[3] = (unsigned)((ga >> 32) & 0x01ffffffu) | (2u << 30); // addr[56:32],type=2
  v8i g1;
  g1[0] = (1 << 16)        // data_size = 2 bytes
        | (1 << 20)        // pad_enable
        | (4 << 22)        // pad_interval: 32 dwords (code 4 -> 2<<4)
        | (3 << 25);       // pad_amount: 4 dwords (code 3 -> 3+1)
  g1[1] = (int)(64u << 16);                 // tensor_dim0 = 64  (lo16 @ [31:16])
  g1[2] = (int)((unsigned)rows << 16);      // tensor_dim1 = rows
  g1[3] = (int)(64u << 16);                 // tile_dim0 = 64    (@ [31:16])
  g1[4] = rows;                             // tile_dim1 = rows  (@ [15:0])
  g1[5] = 512;                              // tensor_dim0_stride lo32
  g1[6] = 0;
  g1[7] = 0;
  const v4i z4 = {0, 0, 0, 0};
  const v8i z8 = {0, 0, 0, 0, 0, 0, 0, 0};
  __builtin_amdgcn_tensor_load_to_lds(g0, g1, z4, z4, z8, 0);
}
#endif

__device__ __forceinline__ v4bf cvt4(float4 x) {
  v4bf p; p[0] = (__bf16)x.x; p[1] = (__bf16)x.y;
  p[2] = (__bf16)x.z; p[3] = (__bf16)x.w; return p;
}

// ---------------------------------------------------------------------------
// Tiled WMMA GEMM: Y[M,N] = X[M,K] * W[K,N] + bias, 64x64 tile, 4 waves/block.
// ---------------------------------------------------------------------------
template <typename TX, bool OUT_BF16>
__global__ __launch_bounds__(128) void wmma_gemm_bias(
    const TX* __restrict__ X, const float* __restrict__ W,
    const float* __restrict__ bias, void* __restrict__ Yv,
    int M, int N, int K) {
  constexpr int LDA = 40, LDB = 72;
  __shared__ alignas(16) __bf16 As[64 * LDA];
  __shared__ alignas(16) __bf16 Bs[32 * LDB];
  const int tid = threadIdx.x, lane = tid & 31, w = tid >> 5;
  const int n0 = blockIdx.x * 64, m0 = blockIdx.y * 64;

  v8f acc[4];
#pragma unroll
  for (int n = 0; n < 4; ++n) acc[n] = (v8f){0.f, 0.f, 0.f, 0.f, 0.f, 0.f, 0.f, 0.f};

  for (int kk = 0; kk < K; kk += 32) {
    if constexpr (sizeof(TX) == 4) {           // fp32 A tile 64x32: cvt->bf16
#pragma unroll
      for (int i = 0; i < 4; ++i) {
        const int f4 = tid + i * 128;          // 512 float4
        const int r = f4 >> 3, c4 = f4 & 7;
        const float4 x = *(const float4*)&X[(size_t)(m0 + r) * K + kk + c4 * 4];
        *(v4bf*)&As[r * LDA + c4 * 4] = cvt4(x);
      }
    } else {                                   // bf16 A tile: raw 16B copies
#pragma unroll
      for (int i = 0; i < 2; ++i) {
        const int u4 = tid + i * 128;          // 256 uint4
        const int r = u4 >> 2, c8 = u4 & 3;
        *(uint4*)&As[r * LDA + c8 * 8] =
            *(const uint4*)&X[(size_t)(m0 + r) * K + kk + c8 * 8];
      }
    }
#pragma unroll
    for (int i = 0; i < 4; ++i) {              // fp32 W tile 32x64: cvt->bf16
      const int f4 = tid + i * 128;            // 512 float4
      const int r = f4 >> 4, c4 = f4 & 15;
      const float4 x = *(const float4*)&W[(size_t)(kk + r) * N + n0 + c4 * 4];
      *(v4bf*)&Bs[r * LDB + c4 * 4] = cvt4(x);
    }
    __syncthreads();
    const v16bf a = ldsA_frag(&As[(w * 16) * LDA], LDA);
#pragma unroll
    for (int n = 0; n < 4; ++n) {
      const v16bf bf = ldsB_frag(&Bs[n * 16], LDB);
      acc[n] = WMMA_BF16(a, bf, acc[n]);
    }
    __syncthreads();
  }

  const int colb  = lane & 15;
  const int rbase = m0 + w * 16 + ((lane < 16) ? 0 : 8);
#pragma unroll
  for (int n = 0; n < 4; ++n) {
    const int col = n0 + n * 16 + colb;
    const float bval = bias[col];
#pragma unroll
    for (int r = 0; r < 8; ++r) {
      const float v = acc[n][r] + bval;
      const size_t off = (size_t)(rbase + r) * N + col;
      if constexpr (OUT_BF16) ((__bf16*)Yv)[off] = (__bf16)v;
      else                    ((float*)Yv)[off]  = v;
    }
  }
}

// ---------------------------------------------------------------------------
// Fused gated attention. Block = (b,h) x 64 query rows; 4 waves x 16 rows.
// ---------------------------------------------------------------------------
__global__ __launch_bounds__(128) void wmma_gating_attn(
    const __bf16* __restrict__ qp, const __bf16* __restrict__ kp,
    const __bf16* __restrict__ vp, __bf16* __restrict__ ao) {
  constexpr int LDQ = 72, LDK = 72, LDV = 72, LDP = 40;
  __shared__ alignas(16) __bf16 Qs[64 * LDQ];
  __shared__ alignas(16) __bf16 Ks[32 * LDK];
  __shared__ alignas(16) __bf16 Vs[32 * LDV];
  __shared__ alignas(16) __bf16 Pb[4 * 16 * LDP];
  __shared__ float muS[64], isS[64];

  const int tid = threadIdx.x, lane = tid & 31, w = tid >> 5;
  const int b = blockIdx.x / Hh, h = blockIdx.x % Hh;
  const int q0 = blockIdx.y * 64;
  const size_t rs   = (size_t)Ncols;                       // 512
  const size_t base = ((size_t)b * Ss) * rs + (size_t)h * Dd;

  // ---- load 64x64 Q tile once (TDM DMA or vectorized fallback) ------------
#if USE_TDM
  if (w == 2) { tdm_load_tile(qp + base + (size_t)q0 * rs, lds_off(Qs), 64); TDM_WAIT(); }
#else
#pragma unroll
  for (int i = 0; i < 4; ++i) {
    const int u4 = tid + i * 128;              // 512 uint4
    const int r = u4 >> 3, c8 = u4 & 7;
    *(uint4*)&Qs[r * LDQ + c8 * 8] =
        *(const uint4*)&qp[base + (size_t)(q0 + r) * rs + c8 * 8];
  }
#endif
  __syncthreads();
  const v16bf aQ0 = ldsA_frag(&Qs[(w * 16) * LDQ + 0], LDQ);
  const v16bf aQ1 = ldsA_frag(&Qs[(w * 16) * LDQ + 32], LDQ);

  float sum_r[8], sq_r[8];
#pragma unroll
  for (int r = 0; r < 8; ++r) { sum_r[r] = 0.f; sq_r[r] = 0.f; }
  const v8f vz = (v8f){0.f, 0.f, 0.f, 0.f, 0.f, 0.f, 0.f, 0.f};

  // ---------------- pass 1: row statistics of logits -----------------------
  for (int sc = 0; sc < Ss; sc += 32) {
#if USE_TDM
    if (w == 0) { tdm_load_tile(kp + base + (size_t)sc * rs, lds_off(Ks), 32); TDM_WAIT(); }
#else
#pragma unroll
    for (int i = 0; i < 2; ++i) {
      const int u4 = tid + i * 128;            // 256 uint4
      const int r = u4 >> 3, c8 = u4 & 7;
      *(uint4*)&Ks[r * LDK + c8 * 8] =
          *(const uint4*)&kp[base + (size_t)(sc + r) * rs + c8 * 8];
    }
#endif
    __syncthreads();
#pragma unroll
    for (int t = 0; t < 2; ++t) {
      v8f s = vz;
      s = WMMA_BF16(aQ0, ldsBT_frag(&Ks[(t * 16) * LDK + 0],  LDK), s);
      s = WMMA_BF16(aQ1, ldsBT_frag(&Ks[(t * 16) * LDK + 32], LDK), s);
#pragma unroll
      for (int r = 0; r < 8; ++r) { const float x = s[r]; sum_r[r] += x; sq_r[r] += x * x; }
    }
    __syncthreads();
  }
#pragma unroll
  for (int r = 0; r < 8; ++r) {
    const float s  = half_reduce(sum_r[r]);
    const float qq = half_reduce(sq_r[r]);
    if ((lane & 15) == 0) {
      const int row = w * 16 + r + ((lane < 16) ? 0 : 8);
      const float mean = s * (1.0f / Ss);
      const float var  = qq * (1.0f / Ss) - mean * mean;
      const float std  = sqrtf(var + 1e-6f) + 1e-6f;
      muS[row] = mean;
      isS[row] = 1.0f / std;
    }
  }
  __syncthreads();
  float mu_r[8], is_r[8];
#pragma unroll
  for (int r = 0; r < 8; ++r) {
    const int row = w * 16 + r + ((lane < 16) ? 0 : 8);
    mu_r[r] = muS[row];
    is_r[r] = isS[row];
  }

  // ---------------- pass 2: softmax num/denominator + P.V ------------------
  v8f O[4];
#pragma unroll
  for (int n = 0; n < 4; ++n) O[n] = vz;
  float den_r[8];
#pragma unroll
  for (int r = 0; r < 8; ++r) den_r[r] = 0.f;
  __bf16* Pw = &Pb[(w * 16) * LDP];

  for (int sc = 0; sc < Ss; sc += 32) {
#if USE_TDM
    if (w == 0) { tdm_load_tile(kp + base + (size_t)sc * rs, lds_off(Ks), 32); TDM_WAIT(); }
    if (w == 1) { tdm_load_tile(vp + base + (size_t)sc * rs, lds_off(Vs), 32); TDM_WAIT(); }
#else
#pragma unroll
    for (int i = 0; i < 2; ++i) {
      const int u4 = tid + i * 128;
      const int r = u4 >> 3, c8 = u4 & 7;
      *(uint4*)&Ks[r * LDK + c8 * 8] =
          *(const uint4*)&kp[base + (size_t)(sc + r) * rs + c8 * 8];
      *(uint4*)&Vs[r * LDV + c8 * 8] =
          *(const uint4*)&vp[base + (size_t)(sc + r) * rs + c8 * 8];
    }
#endif
    __syncthreads();
#pragma unroll
    for (int t = 0; t < 2; ++t) {
      v8f s = vz;
      s = WMMA_BF16(aQ0, ldsBT_frag(&Ks[(t * 16) * LDK + 0],  LDK), s);
      s = WMMA_BF16(aQ1, ldsBT_frag(&Ks[(t * 16) * LDK + 32], LDK), s);
#pragma unroll
      for (int r = 0; r < 8; ++r) {
        const float z = (s[r] - mu_r[r]) * is_r[r];
        const float p = __expf(z);             // normalized logits -> bounded
        den_r[r] += p;
        const int row16 = r + ((lane < 16) ? 0 : 8);
        Pw[row16 * LDP + t * 16 + (lane & 15)] = (__bf16)p;
      }
    }
    const v16bf aP = ldsA_frag(Pw, LDP);       // wave-private, LDS in-order
#pragma unroll
    for (int n = 0; n < 4; ++n)
      O[n] = WMMA_BF16(aP, ldsB_frag(&Vs[n * 16], LDV), O[n]);
    __syncthreads();
  }

#pragma unroll
  for (int r = 0; r < 8; ++r) den_r[r] = half_reduce(den_r[r]);
  const int colb = lane & 15;
#pragma unroll
  for (int n = 0; n < 4; ++n) {
#pragma unroll
    for (int r = 0; r < 8; ++r) {
      const int row = q0 + w * 16 + r + ((lane < 16) ? 0 : 8);
      const float v = O[n][r] / den_r[r];
      ao[base + (size_t)row * rs + n * 16 + colb] = (__bf16)v;
    }
  }
}

// ---------------------------------------------------------------------------
extern "C" void kernel_launch(void* const* d_in, const int* in_sizes, int n_in,
                              void* d_out, int out_size, void* d_ws, size_t ws_size,
                              hipStream_t stream) {
  (void)in_sizes; (void)n_in; (void)out_size; (void)ws_size;
  const float* q  = (const float*)d_in[0];
  const float* k  = (const float*)d_in[1];
  const float* v  = (const float*)d_in[2];
  const float* Wq = (const float*)d_in[3];
  const float* bq = (const float*)d_in[4];
  const float* Wk = (const float*)d_in[5];
  const float* bk = (const float*)d_in[6];
  const float* Wv = (const float*)d_in[7];
  const float* bv = (const float*)d_in[8];
  const float* Wo = (const float*)d_in[9];
  const float* bo = (const float*)d_in[10];

  __bf16* qp = (__bf16*)d_ws;                       // [8192, 512] bf16 each
  __bf16* kp = qp + (size_t)Mrows * Ncols;
  __bf16* vp = kp + (size_t)Mrows * Ncols;
  __bf16* ao = vp + (size_t)Mrows * Ncols;

  const dim3 blk(128);
  const dim3 ggrid(Ncols / 64, Mrows / 64);         // (8, 128)

  wmma_gemm_bias<float, true><<<ggrid, blk, 0, stream>>>(q, Wq, bq, qp, Mrows, Ncols, Ee);
  wmma_gemm_bias<float, true><<<ggrid, blk, 0, stream>>>(k, Wk, bk, kp, Mrows, Ncols, Ee);
  wmma_gemm_bias<float, true><<<ggrid, blk, 0, stream>>>(v, Wv, bv, vp, Mrows, Ncols, Ee);

  const dim3 agrid(Bb * Hh, Ss / 64);               // (32, 32)
  wmma_gating_attn<<<agrid, blk, 0, stream>>>(qp, kp, vp, ao);

  wmma_gemm_bias<__bf16, false><<<ggrid, blk, 0, stream>>>(ao, Wo, bo, d_out, Mrows, Ee, Ncols);
}
